// MoE_36240934043697
// MI455X (gfx1250) — matrix operand order repaired and checked by reading, not compiled, not run on hardware
//
#include <hip/hip_runtime.h>
#include <hip/hip_bf16.h>

// ---------------- problem constants ----------------
#define BATCH 4
#define SEQ   2048
#define DIM   1024      // D
#define EXP   8         // E
#define FFN   4096      // F
#define NTOK  (BATCH*SEQ)   // 8192 tokens

// ---------------- tiling ----------------
#define MT    64                 // tokens per workgroup tile (4 x 16-row quarters)
#define FC    256                // F chunk per iteration (16 strips x 16 cols)
#define TILES_PER_E (NTOK / MT)  // 128
#define XS_STRIDE (DIM + 8)      // bf16 elems; 2064B row stride -> conflict-free b128 reads
#define HS_STRIDE (FC + 8)       // bf16 elems; 528B row stride  -> conflict-free b128 reads

typedef __attribute__((ext_vector_type(16))) __bf16 v16bf;
typedef __attribute__((ext_vector_type(8)))  __bf16 v8bf;
typedef __attribute__((ext_vector_type(4)))  __bf16 v4bf;
typedef __attribute__((ext_vector_type(8)))  float  v8f;

// ---------------- helpers ----------------
// Lowers to v_cvt_pk_bf16_f32 on gfx1250 (confirmed in disasm).
__device__ __forceinline__ __bf16 f2bf(float f) { return (__bf16)f; }

// Branch-free tanh-GELU: 0.5*v*(1+tanh(c*(v+0.044715 v^3))) == v * sigmoid(2c*(v+0.044715 v^3))
//   = v * rcp(1 + exp2(-log2(e) * 2c * (v + 0.044715 v^3)))
// Straight-line: ~8 VALU (2 trans), no EXEC divergence (vs libm tanhf's branchy slow path).
__device__ __forceinline__ float gelu_fast(float v) {
    float vv = v * v;
    float inner = v * __builtin_fmaf(0.044715f, vv, 1.0f);   // v + 0.044715 v^3
    // -2*c*log2(e), c = sqrt(2/pi)=0.7978845608 -> -2.302118131
    float t = __builtin_amdgcn_exp2f(inner * -2.302118131f);
    return v * __builtin_amdgcn_rcpf(1.0f + t);
}

__device__ __forceinline__ v8f wmma_bf16(v16bf a, v16bf b, v8f c) {
    return __builtin_amdgcn_wmma_f32_16x16x32_bf16(
        /*neg_a=*/false, a, /*neg_b=*/false, b,
        /*c_mod=*/(short)0, c, /*reuse_a=*/false, /*reuse_b=*/false);
}

// A-matrix (16x32 bf16, MxK) from row-major LDS.
// lane<16:  elems 0..15 = K[kc..kc+15],    row M=lane
// lane>=16: elems 0..15 = K[kc+16..kc+31], row M=lane-16
__device__ __forceinline__ v16bf load_A_lds(const __bf16* base, int stride,
                                            int mbase, int kc, int lane) {
    int grp = lane >> 4;
    int m   = mbase + (lane & 15);
    const __bf16* rp = base + m * stride + kc + grp * 16;
    v8bf lo = *(const v8bf*)(rp);
    v8bf hi = *(const v8bf*)(rp + 8);
    v16bf a;
#pragma unroll
    for (int i = 0; i < 8; ++i) { a[i] = lo[i]; a[i + 8] = hi[i]; }
    return a;
}

// B-matrix (32x16 bf16, KxN) from row-major fp32 global W[ld = row length].
// lane<16: elems j = K[krow0+j], col N=lane;  lane>=16: K[krow0+16+j], col N=lane-16
// Consecutive j pairs pack into one dword -> v_cvt_pk_bf16_f32 friendly.
template <int LD>
__device__ __forceinline__ v16bf load_B_global(const float* __restrict__ W,
                                               int kbase, int nbase, int lane) {
    int col   = nbase + (lane & 15);
    int krow0 = kbase + ((lane >> 4) << 4);
    const float* p = W + (size_t)krow0 * LD + col;
    v16bf b;
#pragma unroll
    for (int j = 0; j < 16; ++j)
        b[j] = f2bf(p[(size_t)j * LD]);
    return b;
}

// ---------------- router: logits -> top2 -> softmax -> bucket append ----------------
__global__ __launch_bounds__(256) void moe_router_kernel(
    const float* __restrict__ x, const float* __restrict__ rw,
    int* __restrict__ counts, int* __restrict__ tok_ids, float* __restrict__ gates) {
    int wave = blockIdx.x * (blockDim.x >> 5) + (threadIdx.x >> 5);
    int lane = threadIdx.x & 31;
    if (wave >= NTOK) return;
    const float* xr = x + (size_t)wave * DIM;
    float p[EXP];
#pragma unroll
    for (int e = 0; e < EXP; ++e) p[e] = 0.0f;
    for (int d = lane; d < DIM; d += 32) {
        float xv = xr[d];
#pragma unroll
        for (int e = 0; e < EXP; ++e) p[e] += xv * rw[d * EXP + e];
    }
#pragma unroll
    for (int e = 0; e < EXP; ++e) {
#pragma unroll
        for (int off = 16; off > 0; off >>= 1) p[e] += __shfl_xor(p[e], off, 32);
    }
    if (lane == 0) {
        int e1 = 0; float m1 = p[0];
#pragma unroll
        for (int e = 1; e < EXP; ++e) if (p[e] > m1) { m1 = p[e]; e1 = e; }
        int e2 = -1; float m2 = -INFINITY;
#pragma unroll
        for (int e = 0; e < EXP; ++e) if (e != e1 && p[e] > m2) { m2 = p[e]; e2 = e; }
        // softmax over {m1, m2}: q = e^(m2-m1) <= 1, branch-free native exp2/rcp
        float q  = __builtin_amdgcn_exp2f((m2 - m1) * 1.4426950408889634f);
        float r  = __builtin_amdgcn_rcpf(1.0f + q);
        float g1 = r;
        float g2 = q * r;
        int s1 = atomicAdd(&counts[e1], 1);
        tok_ids[e1 * NTOK + s1] = wave; gates[e1 * NTOK + s1] = g1;
        int s2 = atomicAdd(&counts[e2], 1);
        tok_ids[e2 * NTOK + s2] = wave; gates[e2 * NTOK + s2] = g2;
    }
}

// ---------------- fused expert FFN: out += g * (gelu(X@w1+b1) @ w2 + b2) ----------------
// 512 threads = 16 waves. Each wave processes all FOUR 16-token quarters (mi=0..3) so
// every converted weight B-tile feeds 4 WMMAs.
//  phase 1: wave = F strip fj=w (16 cols of the 256-wide chunk), 4 accumulators
//  phase 2: wave = D strip dj=w (64 cols), 4x4 accumulators (full 64x1024 Y in WG regs)
__global__ __launch_bounds__(512) void moe_expert_kernel(
    const float* __restrict__ x,
    const float* __restrict__ w1, const float* __restrict__ b1,
    const float* __restrict__ w2, const float* __restrict__ b2,
    const int* __restrict__ counts, const int* __restrict__ tok_ids,
    const float* __restrict__ gates, float* __restrict__ out) {

    __shared__ __align__(16) __bf16 Xs[MT * XS_STRIDE];  // 132096 B
    __shared__ __align__(16) __bf16 Hs[MT * HS_STRIDE];  //  33792 B
    __shared__ int   tok_s[MT];
    __shared__ float gate_s[MT];

    const int e    = blockIdx.x / TILES_PER_E;
    const int tile = blockIdx.x % TILES_PER_E;
    const int ne   = counts[e];
    const int base = tile * MT;
    if (base >= ne) return;

    const int tid  = threadIdx.x;
    const int w    = tid >> 5;        // wave 0..15
    const int lane = tid & 31;
    const int grp  = lane >> 4;
    const int ln   = lane & 15;

    // stage token ids + gates
    if (tid < MT) {
        int idx = base + tid;
        int t = 0; float g = 0.0f;
        if (idx < ne) { t = tok_ids[e * NTOK + idx]; g = gates[e * NTOK + idx]; }
        tok_s[tid] = t; gate_s[tid] = g;
    }
    __syncthreads();

    // stage X tile (gathered rows) as bf16, vectorized b128 -> 2x cvt_pk -> b64
    for (int i = tid * 4; i < MT * DIM; i += (int)blockDim.x * 4) {
        int row = i >> 10;            // /DIM
        int col = i & (DIM - 1);
        const float4 xv = *(const float4*)(x + (size_t)tok_s[row] * DIM + col);
        v4bf hv;
        hv[0] = f2bf(xv.x); hv[1] = f2bf(xv.y); hv[2] = f2bf(xv.z); hv[3] = f2bf(xv.w);
        *(v4bf*)(&Xs[row * XS_STRIDE + col]) = hv;
    }
    __syncthreads();

    const float* W1e = w1 + (size_t)e * DIM * FFN;
    const float* W2e = w2 + (size_t)e * FFN * DIM;
    const float* b1e = b1 + (size_t)e * FFN;
    const float* b2e = b2 + (size_t)e * DIM;

    v8f acc[4][4];                    // persistent 64x64 f32 Y slice per wave (128 VGPRs)
#pragma unroll
    for (int mi = 0; mi < 4; ++mi)
#pragma unroll
        for (int nt = 0; nt < 4; ++nt)
            acc[mi][nt] = (v8f){0.f,0.f,0.f,0.f,0.f,0.f,0.f,0.f};

    for (int fc = 0; fc < FFN; fc += FC) {
        // ---- phase 1: H[64 x FC] = gelu(X @ w1 + b1); wave owns 16-col strip, 4 quarters
        const int ncol = fc + w * 16;
        v8f hc[4];
#pragma unroll
        for (int mi = 0; mi < 4; ++mi)
            hc[mi] = (v8f){0.f,0.f,0.f,0.f,0.f,0.f,0.f,0.f};
        for (int kc = 0; kc < DIM; kc += 32) {
            v16bf b = load_B_global<FFN>(W1e, kc, ncol, lane);    // loaded once...
#pragma unroll
            for (int mi = 0; mi < 4; ++mi) {
                v16bf a = load_A_lds(Xs, XS_STRIDE, mi * 16, kc, lane);
                hc[mi] = wmma_bf16(a, b, hc[mi]);                 // ...used 4 times
            }
        }
        float b1c = b1e[ncol + ln];
#pragma unroll
        for (int mi = 0; mi < 4; ++mi) {
#pragma unroll
            for (int j = 0; j < 8; ++j) {
                // C layout: VGPR j -> M=j (lanes 0-15), M=j+8 (lanes 16-31)
                int r = mi * 16 + j + grp * 8;
                Hs[r * HS_STRIDE + w * 16 + ln] = f2bf(gelu_fast(hc[mi][j] + b1c));
            }
        }
        __syncthreads();

        // ---- phase 2: acc += H @ w2 ; wave owns 64-col D strip, 4 quarters
#pragma unroll 2
        for (int kc = 0; kc < FC; kc += 32) {
            v16bf a0 = load_A_lds(Hs, HS_STRIDE,  0, kc, lane);
            v16bf a1 = load_A_lds(Hs, HS_STRIDE, 16, kc, lane);
            v16bf a2 = load_A_lds(Hs, HS_STRIDE, 32, kc, lane);
            v16bf a3 = load_A_lds(Hs, HS_STRIDE, 48, kc, lane);
#pragma unroll
            for (int nt = 0; nt < 4; ++nt) {
                v16bf b = load_B_global<DIM>(W2e, fc + kc, w * 64 + nt * 16, lane);
                acc[0][nt] = wmma_bf16(a0, b, acc[0][nt]);
                acc[1][nt] = wmma_bf16(a1, b, acc[1][nt]);
                acc[2][nt] = wmma_bf16(a2, b, acc[2][nt]);
                acc[3][nt] = wmma_bf16(a3, b, acc[3][nt]);
            }
        }
        __syncthreads();   // before Hs is overwritten next chunk
    }

    // ---- epilogue: out[t] += gate * (acc + b2)
#pragma unroll
    for (int mi = 0; mi < 4; ++mi) {
#pragma unroll
        for (int nt = 0; nt < 4; ++nt) {
            int coln = w * 64 + nt * 16 + ln;
            float b2c = b2e[coln];
#pragma unroll
            for (int j = 0; j < 8; ++j) {
                int r = mi * 16 + j + grp * 8;
                float val = gate_s[r] * (acc[mi][nt][j] + b2c);
                float* dst = out + (size_t)tok_s[r] * DIM + coln;
                __hip_atomic_fetch_add(dst, val, __ATOMIC_RELAXED, __HIP_MEMORY_SCOPE_AGENT);
            }
        }
    }
}

// ---------------- launch ----------------
extern "C" void kernel_launch(void* const* d_in, const int* in_sizes, int n_in,
                              void* d_out, int out_size, void* d_ws, size_t ws_size,
                              hipStream_t stream) {
    const float* x   = (const float*)d_in[0];
    const float* rw  = (const float*)d_in[1];
    const float* w1  = (const float*)d_in[2];
    const float* b1  = (const float*)d_in[3];
    const float* w2  = (const float*)d_in[4];
    const float* b2  = (const float*)d_in[5];
    float* out = (float*)d_out;

    // workspace: [counts: 256B pad][tok_ids: E*NTOK int][gates: E*NTOK float]
    int*   counts  = (int*)d_ws;
    int*   tok_ids = (int*)((char*)d_ws + 256);
    float* gates   = (float*)((char*)d_ws + 256 + (size_t)EXP * NTOK * sizeof(int));

    hipMemsetAsync(d_ws, 0, 256, stream);
    hipMemsetAsync(d_out, 0, (size_t)out_size * sizeof(float), stream);

    moe_router_kernel<<<NTOK / 8, 256, 0, stream>>>(x, rw, counts, tok_ids, gates);
    moe_expert_kernel<<<EXP * TILES_PER_E, 512, 0, stream>>>(
        x, w1, b1, w2, b2, counts, tok_ids, gates, out);
}
